// GnReluFinefy_25400436588659
// MI455X (gfx1250) — matrix operand order repaired and checked by reading, not compiled, not run on hardware
//
#include <hip/hip_runtime.h>
#include <hip/hip_bf16.h>

typedef __attribute__((ext_vector_type(16))) _Float16 v16h;
typedef __attribute__((ext_vector_type(8)))  _Float16 v8h;
typedef __attribute__((ext_vector_type(8)))  float    v8f;

#define N_COARSE 100000
#define N_FINE   300000
#define VAL_DIM  64
#define FEXT     9
#define NFILT    64
#define KDIM     (FEXT * VAL_DIM)   // 576
#define GN_EPS   1e-5f
#define LDS_STRIDE 584              // 576 + 8 halves pad -> conflict-free B reads
#define TILES    (N_FINE / 16)      // 18750, exact
#define WAVES_PER_BLOCK 8

// ---------------------------------------------------------------------------
// ws layout (bytes):
//   [0,   512)                     : float stats -> sum[64], sumsq[64]
//   [1024, 1024 + 12,800,000)      : xhat, f16 [N_COARSE][64]  (normalized+relu)
//   [then, +73,728)                : wt,   f16 [64][576]       (weight^T)
// ---------------------------------------------------------------------------

__global__ void zero_stats_kernel(float* __restrict__ stats) {
    if (threadIdx.x < 128) stats[threadIdx.x] = 0.0f;
}

// Per-channel sum / sum-of-squares over all N_COARSE rows.
// 256 threads: channel c = tid&63, row-subgroup = tid>>6 (4 row streams).
__global__ __launch_bounds__(256) void gn_stats_kernel(
    const float* __restrict__ x, float* __restrict__ stats) {
    __shared__ float sh_s[256];
    __shared__ float sh_q[256];
    const int tid = threadIdx.x;
    const int c   = tid & 63;
    const int rg  = tid >> 6;
    float s = 0.0f, q = 0.0f;
    for (int r = blockIdx.x * 4 + rg; r < N_COARSE; r += gridDim.x * 4) {
        const float v = x[(long long)r * VAL_DIM + c];
        s += v;
        q += v * v;
    }
    sh_s[tid] = s;
    sh_q[tid] = q;
    __syncthreads();
    if (tid < 64) {
        #pragma unroll
        for (int i = 1; i < 4; ++i) {
            s += sh_s[tid + 64 * i];
            q += sh_q[tid + 64 * i];
        }
        atomicAdd(&stats[c], s);
        atomicAdd(&stats[64 + c], q);
    }
}

// GroupNorm (32 groups of 2 channels over all rows) + affine + ReLU -> f16.
__global__ __launch_bounds__(256) void gn_normalize_kernel(
    const float* __restrict__ x, const float* __restrict__ stats,
    const float* __restrict__ gamma, const float* __restrict__ beta,
    _Float16* __restrict__ xhat) {
    const long long i = (long long)blockIdx.x * 256 + threadIdx.x;
    const int c = (int)(i & 63);
    const int g = c >> 1;
    const float cnt  = 2.0f * (float)N_COARSE;
    const float s    = stats[2 * g] + stats[2 * g + 1];
    const float sq   = stats[64 + 2 * g] + stats[64 + 2 * g + 1];
    const float mean = s / cnt;
    const float var  = sq / cnt - mean * mean;
    const float rstd = rsqrtf(var + GN_EPS);
    float y = (x[i] - mean) * rstd * gamma[c] + beta[c];
    y = fmaxf(y, 0.0f);
    xhat[i] = (_Float16)y;
}

// weight [576][64] f32  ->  wt [64][576] f16 (B-major: row = output filter n).
__global__ __launch_bounds__(256) void weight_transpose_kernel(
    const float* __restrict__ w, _Float16* __restrict__ wt) {
    const int i = blockIdx.x * 256 + threadIdx.x;   // over 64*576
    const int n = i / KDIM;
    const int k = i % KDIM;
    wt[i] = (_Float16)w[k * NFILT + n];
}

// Gather + GEMM via WMMA. One wave = one 16-row fine tile, full K=576,
// all 64 output filters (4 accumulators of 16x16 f32).
__global__ __launch_bounds__(256) void finefy_wmma_kernel(
    const _Float16* __restrict__ xhat,      // [N_COARSE][64] f16
    const _Float16* __restrict__ wt,        // [64][576] f16
    const long long* __restrict__ nidx,     // [N_FINE][9] int64
    float* __restrict__ out) {              // [N_FINE][64] f32
    __shared__ __align__(16) _Float16 wlds[64 * LDS_STRIDE];

    const int tid = threadIdx.x;

    // Cooperative weight load global -> LDS (72 x 16B vectors per row).
    for (int q = tid; q < 64 * 72; q += 256) {
        const int r = q / 72;
        const int o = q % 72;
        const uint4 v = ((const uint4*)wt)[r * 72 + o];
        *(uint4*)&wlds[r * LDS_STRIDE + o * 8] = v;
    }
    __syncthreads();

    const int wave = tid >> 5;
    const int lane = tid & 31;
    const int tile = blockIdx.x * WAVES_PER_BLOCK + wave;
    if (tile >= TILES) return;

    const int row0 = tile * 16;
    const int mrow = lane & 15;       // A row / C column-lane
    const int hi   = lane >> 4;       // half-wave select (K sub-block / M half)
    const int row  = row0 + mrow;

    // Preload the 9 neighbor indices for this lane's A-row.
    long long nb[FEXT];
    #pragma unroll
    for (int f = 0; f < FEXT; ++f) nb[f] = nidx[(long long)row * FEXT + f];

    v8f acc0 = {}, acc1 = {}, acc2 = {}, acc3 = {};

    // K walked in 18 chunks of 32: chunk t -> neighbor f=t>>1, channels (t&1)*32..+31.
    #pragma unroll
    for (int t = 0; t < 18; ++t) {
        const int f     = t >> 1;
        const int choff = (t & 1) * 32;

        // A fragment: lane holds K = base+0..15 (contiguous, matches B),
        // base = hi*16, within this 32-wide chunk. Two contiguous 16B loads.
        const _Float16* src = xhat + nb[f] * VAL_DIM + choff + hi * 16;
        const v8h alo = *(const v8h*)(src);
        const v8h ahi = *(const v8h*)(src + 8);
        const v16h a = __builtin_shufflevector(alo, ahi,
            0, 1, 2, 3, 4, 5, 6, 7, 8, 9, 10, 11, 12, 13, 14, 15);

        // B fragments (contiguous K order, same as A): lane n = lane&15, contiguous K halves
        // starting at t*32 + hi*16. One per 16-wide N tile.
        const int kb = t * 32 + hi * 16;
        const _Float16* bp = &wlds[(lane & 15) * LDS_STRIDE + kb];
        #define LOAD_B(ofs) __builtin_shufflevector(                          \
            *(const v8h*)(bp + (ofs)), *(const v8h*)(bp + (ofs) + 8),         \
            0, 1, 2, 3, 4, 5, 6, 7, 8, 9, 10, 11, 12, 13, 14, 15)
        const v16h b0 = LOAD_B(0 * 16 * LDS_STRIDE);
        const v16h b1 = LOAD_B(1 * 16 * LDS_STRIDE);
        const v16h b2 = LOAD_B(2 * 16 * LDS_STRIDE);
        const v16h b3 = LOAD_B(3 * 16 * LDS_STRIDE);
        #undef LOAD_B

        acc0 = __builtin_amdgcn_wmma_f32_16x16x32_f16(false, a, false, b0, (short)0, acc0, false, false);
        acc1 = __builtin_amdgcn_wmma_f32_16x16x32_f16(false, a, false, b1, (short)0, acc1, false, false);
        acc2 = __builtin_amdgcn_wmma_f32_16x16x32_f16(false, a, false, b2, (short)0, acc2, false, false);
        acc3 = __builtin_amdgcn_wmma_f32_16x16x32_f16(false, a, false, b3, (short)0, acc3, false, false);
    }

    // C/D layout: VGPR r -> M = r + 8*hi, N = ntile*16 + (lane&15).
    const int  ncol = lane & 15;
    float* orow = out + (long long)row0 * NFILT;
    #pragma unroll
    for (int r = 0; r < 8; ++r) {
        const int m = hi * 8 + r;
        float* p = orow + (long long)m * NFILT + ncol;
        p[0]  = acc0[r];
        p[16] = acc1[r];
        p[32] = acc2[r];
        p[48] = acc3[r];
    }
}

extern "C" void kernel_launch(void* const* d_in, const int* in_sizes, int n_in,
                              void* d_out, int out_size, void* d_ws, size_t ws_size,
                              hipStream_t stream) {
    const float*     lv     = (const float*)d_in[0];
    const long long* nidx   = (const long long*)d_in[1];
    const float*     gamma  = (const float*)d_in[2];
    const float*     beta   = (const float*)d_in[3];
    const float*     weight = (const float*)d_in[4];
    float*           out    = (float*)d_out;

    char* ws = (char*)d_ws;
    float*     stats = (float*)ws;
    _Float16*  xhat  = (_Float16*)(ws + 1024);
    _Float16*  wt    = (_Float16*)(ws + 1024 + (size_t)N_COARSE * VAL_DIM * sizeof(_Float16));

    zero_stats_kernel<<<1, 128, 0, stream>>>(stats);
    gn_stats_kernel<<<512, 256, 0, stream>>>(lv, stats);
    gn_normalize_kernel<<<(N_COARSE * VAL_DIM) / 256, 256, 0, stream>>>(lv, stats, gamma, beta, xhat);
    weight_transpose_kernel<<<(KDIM * NFILT) / 256, 256, 0, stream>>>(weight, wt);

    const int blocks = (TILES + WAVES_PER_BLOCK - 1) / WAVES_PER_BLOCK;  // 2344
    finefy_wmma_kernel<<<blocks, 256, 0, stream>>>(xhat, wt, nidx, out);
}
